// MultiHeadAttention_47064251629900
// MI455X (gfx1250) — matrix operand order repaired and checked by reading, not compiled, not run on hardware
//
#include <hip/hip_runtime.h>
#include <hip/hip_bf16.h>

// ---------------------------------------------------------------------------
// CDNA5 (gfx1250) causal multi-head attention, bf16 WMMA, fp32 accumulate.
//   B=1, S=4096, D=2048, H=16, hd=128
// Pipeline:
//   1) gemm_proj : X(f32) x {Wq,Wk,Wv}(f32) -> Q/K/V bf16 head-major [H][S][128]
//                  (one launch, gridDim.z=3; register-staged double buffer)
//   2) attn_flash: online-softmax causal attention -> O bf16 [S][D]
//   3) gemm_out  : O(bf16) x Wo(f32) -> fp32 d_out [S][D]
// ---------------------------------------------------------------------------

typedef __attribute__((ext_vector_type(16))) __bf16 v16bf;
typedef __attribute__((ext_vector_type(8)))  __bf16 v8bf;
typedef __attribute__((ext_vector_type(8)))  float  v8f;

static __device__ __forceinline__ v16bf cat16(v8bf lo, v8bf hi) {
  return __builtin_shufflevector(lo, hi, 0,1,2,3,4,5,6,7,8,9,10,11,12,13,14,15);
}

static __device__ __forceinline__ v8f wmma_bf16(v16bf a, v16bf b, v8f c) {
  // D = A(16x32) * B(32x16) + C(16x16), f32 accumulate
  return __builtin_amdgcn_wmma_f32_16x16x32_bf16(false, a, false, b,
                                                 (short)0, c, false, false);
}

static __device__ __forceinline__ v8bf cvt8(float4 x, float4 y) {
  v8bf o;
  o[0] = (__bf16)x.x; o[1] = (__bf16)x.y; o[2] = (__bf16)x.z; o[3] = (__bf16)x.w;
  o[4] = (__bf16)y.x; o[5] = (__bf16)y.y; o[6] = (__bf16)y.z; o[7] = (__bf16)y.w;
  return o;
}

// ---------------------------------------------------------------------------
// Projection GEMM: C = X[M,K] * W[K,N], X,W fp32, out bf16 head-major
// [H][M][128] (H = N/128). gridDim = (N/128, M/128, 3); z picks Wq/Wk/Wv.
// Block: 256 threads = 8 waves (2x4); tile 128x128x64; wave tile 64x32 (4x2).
// Register-staged double buffer: next global tile loads overlap the WMMAs.
// ---------------------------------------------------------------------------
__global__ __launch_bounds__(256)
void gemm_proj(const float* __restrict__ A,
               const float* __restrict__ W0, const float* __restrict__ W1,
               const float* __restrict__ W2,
               __bf16* __restrict__ D0, __bf16* __restrict__ D1,
               __bf16* __restrict__ D2,
               int M, int N, int K) {
  __shared__ __align__(16) __bf16 sA[128 * 64];   // [m][k]
  __shared__ __align__(16) __bf16 sBt[128 * 64];  // [n][k] (W transposed)

  const int z = blockIdx.z;
  const float* B   = (z == 0) ? W0 : (z == 1) ? W1 : W2;
  __bf16*      dst = (z == 0) ? D0 : (z == 1) ? D1 : D2;

  const int tid   = threadIdx.x;
  const int wave  = tid >> 5;
  const int lane  = tid & 31;
  const int lhalf = lane >> 4;
  const int l16   = lane & 15;
  const int wm    = wave >> 2;
  const int wn    = wave & 3;
  const int m0    = blockIdx.y * 128;
  const int n0    = blockIdx.x * 128;

  v8f acc[4][2] = {};
  float4 a0[4], a1[4], b0[4], b1[4];   // staged global data (raw f32)

  auto gload = [&](int k0) {
#pragma unroll
    for (int p = 0; p < 4; ++p) {
      const int lin = (p * 256 + tid) * 8;
      const int r = lin >> 6, c = lin & 63;        // A tile 128x64
      const float* ga = A + (size_t)(m0 + r) * K + k0 + c;
      a0[p] = *(const float4*)ga;
      a1[p] = *(const float4*)(ga + 4);
      const int kk = lin >> 7, nn = lin & 127;     // B tile 64x128
      const float* gb = B + (size_t)(k0 + kk) * N + n0 + nn;
      b0[p] = *(const float4*)gb;
      b1[p] = *(const float4*)(gb + 4);
    }
  };
  auto lstore = [&]() {
#pragma unroll
    for (int p = 0; p < 4; ++p) {
      const int lin = (p * 256 + tid) * 8;
      *(v8bf*)(sA + lin) = cvt8(a0[p], a1[p]);
      const int kk = lin >> 7, nn = lin & 127;
      v8bf w = cvt8(b0[p], b1[p]);
#pragma unroll
      for (int j = 0; j < 8; ++j) sBt[(nn + j) * 64 + kk] = w[j];
    }
  };

  gload(0);
  for (int k0 = 0; k0 < K; k0 += 64) {
    lstore();
    __syncthreads();
    if (k0 + 64 < K) gload(k0 + 64);   // in flight during WMMAs below

#pragma unroll
    for (int ks = 0; ks < 2; ++ks) {
      const int kb = ks * 32;
      v16bf af[4], bfr[2];
#pragma unroll
      for (int ms = 0; ms < 4; ++ms) {
        const __bf16* base = sA + (wm * 64 + ms * 16 + l16) * 64 + kb + lhalf * 8;
        af[ms] = cat16(*(const v8bf*)base, *(const v8bf*)(base + 16));
      }
#pragma unroll
      for (int ns = 0; ns < 2; ++ns) {
        const __bf16* base = sBt + (wn * 32 + ns * 16 + l16) * 64 + kb + lhalf * 8;
        bfr[ns] = cat16(*(const v8bf*)base, *(const v8bf*)(base + 16));
      }
#pragma unroll
      for (int ms = 0; ms < 4; ++ms)
#pragma unroll
        for (int ns = 0; ns < 2; ++ns)
          acc[ms][ns] = wmma_bf16(af[ms], bfr[ns], acc[ms][ns]);
    }
    __syncthreads();
  }

#pragma unroll
  for (int ms = 0; ms < 4; ++ms)
#pragma unroll
    for (int ns = 0; ns < 2; ++ns) {
      const int col = n0 + wn * 32 + ns * 16 + l16;
      const int h = col >> 7, c = col & 127;
#pragma unroll
      for (int j = 0; j < 8; ++j) {
        const int row = m0 + wm * 64 + ms * 16 + j + 8 * lhalf;
        dst[((size_t)h * M + row) * 128 + c] = (__bf16)acc[ms][ns][j];
      }
    }
}

// ---------------------------------------------------------------------------
// Output GEMM: C = O[M,K](bf16) * Wo[K,N](f32) -> fp32 row-major.
// ---------------------------------------------------------------------------
__global__ __launch_bounds__(256)
void gemm_out(const __bf16* __restrict__ A, const float* __restrict__ B,
              float* __restrict__ dst, int M, int N, int K) {
  __shared__ __align__(16) __bf16 sA[128 * 64];
  __shared__ __align__(16) __bf16 sBt[128 * 64];

  const int tid   = threadIdx.x;
  const int wave  = tid >> 5;
  const int lane  = tid & 31;
  const int lhalf = lane >> 4;
  const int l16   = lane & 15;
  const int wm    = wave >> 2;
  const int wn    = wave & 3;
  const int m0    = blockIdx.y * 128;
  const int n0    = blockIdx.x * 128;

  v8f acc[4][2] = {};
  v8bf   a8[4];
  float4 b0[4], b1[4];

  auto gload = [&](int k0) {
#pragma unroll
    for (int p = 0; p < 4; ++p) {
      const int lin = (p * 256 + tid) * 8;
      const int r = lin >> 6, c = lin & 63;
      a8[p] = *(const v8bf*)(A + (size_t)(m0 + r) * K + k0 + c);
      const int kk = lin >> 7, nn = lin & 127;
      const float* gb = B + (size_t)(k0 + kk) * N + n0 + nn;
      b0[p] = *(const float4*)gb;
      b1[p] = *(const float4*)(gb + 4);
    }
  };
  auto lstore = [&]() {
#pragma unroll
    for (int p = 0; p < 4; ++p) {
      const int lin = (p * 256 + tid) * 8;
      *(v8bf*)(sA + lin) = a8[p];
      const int kk = lin >> 7, nn = lin & 127;
      v8bf w = cvt8(b0[p], b1[p]);
#pragma unroll
      for (int j = 0; j < 8; ++j) sBt[(nn + j) * 64 + kk] = w[j];
    }
  };

  gload(0);
  for (int k0 = 0; k0 < K; k0 += 64) {
    lstore();
    __syncthreads();
    if (k0 + 64 < K) gload(k0 + 64);

#pragma unroll
    for (int ks = 0; ks < 2; ++ks) {
      const int kb = ks * 32;
      v16bf af[4], bfr[2];
#pragma unroll
      for (int ms = 0; ms < 4; ++ms) {
        const __bf16* base = sA + (wm * 64 + ms * 16 + l16) * 64 + kb + lhalf * 8;
        af[ms] = cat16(*(const v8bf*)base, *(const v8bf*)(base + 16));
      }
#pragma unroll
      for (int ns = 0; ns < 2; ++ns) {
        const __bf16* base = sBt + (wn * 32 + ns * 16 + l16) * 64 + kb + lhalf * 8;
        bfr[ns] = cat16(*(const v8bf*)base, *(const v8bf*)(base + 16));
      }
#pragma unroll
      for (int ms = 0; ms < 4; ++ms)
#pragma unroll
        for (int ns = 0; ns < 2; ++ns)
          acc[ms][ns] = wmma_bf16(af[ms], bfr[ns], acc[ms][ns]);
    }
    __syncthreads();
  }

#pragma unroll
  for (int ms = 0; ms < 4; ++ms)
#pragma unroll
    for (int ns = 0; ns < 2; ++ns) {
      const int col = n0 + wn * 32 + ns * 16 + l16;
#pragma unroll
      for (int j = 0; j < 8; ++j) {
        const int row = m0 + wm * 64 + ms * 16 + j + 8 * lhalf;
        dst[(size_t)row * N + col] = acc[ms][ns][j];
      }
    }
}

// ---------------------------------------------------------------------------
// Flash attention: one head / one 64-row query tile per block.
// Block: 128 threads = 4 waves; wave w owns q-rows [w*16, w*16+16).
// Grid: (S/64, H). Online softmax, causal; prefetch next KV tile.
// ---------------------------------------------------------------------------
__global__ __launch_bounds__(128)
void attn_flash(const __bf16* __restrict__ Qh, const __bf16* __restrict__ Kh,
                const __bf16* __restrict__ Vh, __bf16* __restrict__ Obf,
                int S, float scale) {
  constexpr int HD = 128;
  __shared__ __align__(16) __bf16 sQ[64 * HD];    // [q][hd]
  __shared__ __align__(16) __bf16 sK[64 * HD];    // [kv][hd] (== B-frag layout)
  __shared__ __align__(16) __bf16 sVt[HD * 64];   // [hd][kv] (transposed V)
  __shared__ __align__(16) __bf16 sP[64 * 64];    // probs bf16

  const int tid   = threadIdx.x;
  const int wave  = tid >> 5;
  const int lane  = tid & 31;
  const int lhalf = lane >> 4;
  const int l16   = lane & 15;
  const int qbase = blockIdx.x * 64;
  const int h     = blockIdx.y;

  const __bf16* Qp = Qh + ((size_t)h * S + qbase) * HD;
  const __bf16* Kp = Kh + (size_t)h * S * HD;
  const __bf16* Vp = Vh + (size_t)h * S * HD;

#pragma unroll
  for (int p = 0; p < 8; ++p) {
    const int linear = (p * 128 + tid) * 8;
    *(v8bf*)(sQ + linear) = *(const v8bf*)(Qp + linear);
  }
  __syncthreads();

  v16bf qf[4];
#pragma unroll
  for (int ks = 0; ks < 4; ++ks) {
    const __bf16* base = sQ + (wave * 16 + l16) * HD + ks * 32 + lhalf * 8;
    qf[ks] = cat16(*(const v8bf*)base, *(const v8bf*)(base + 16));
  }

  float mrow[8], lrow[8];
#pragma unroll
  for (int j = 0; j < 8; ++j) { mrow[j] = -1e30f; lrow[j] = 0.f; }
  v8f acco[8] = {};

  for (int t = 0; t <= (int)blockIdx.x; ++t) {
    const int kbase = t * 64;
#pragma unroll
    for (int p = 0; p < 8; ++p) {
      const int linear = (p * 128 + tid) * 8;
      *(v8bf*)(sK + linear) = *(const v8bf*)(Kp + (size_t)kbase * HD + linear);
    }
#pragma unroll
    for (int p = 0; p < 8; ++p) {
      const int linear = (p * 128 + tid) * 8;
      const int r = linear >> 7, c = linear & 127;
      v8bf v = *(const v8bf*)(Vp + (size_t)(kbase + r) * HD + c);
#pragma unroll
      for (int j = 0; j < 8; ++j) sVt[(c + j) * 64 + r] = v[j];
    }
    __syncthreads();

    // prefetch next causal KV tile while this one is consumed
    if (t < (int)blockIdx.x) {
      const __bf16* nk = Kp + (size_t)(kbase + 64) * HD;
      const __bf16* nv = Vp + (size_t)(kbase + 64) * HD;
#pragma unroll
      for (int p = 0; p < 8; ++p) {
        const int linear = (p * 128 + tid) * 8;
        __builtin_prefetch(nk + linear, 0, 0);
        __builtin_prefetch(nv + linear, 0, 0);
      }
    }

    // ---- scores S = Q K^T ----
    v8f accs[4] = {};
#pragma unroll
    for (int ks = 0; ks < 4; ++ks)
#pragma unroll
      for (int nt = 0; nt < 4; ++nt) {
        const __bf16* base = sK + (nt * 16 + l16) * HD + ks * 32 + lhalf * 8;
        v16bf bfrag = cat16(*(const v8bf*)base, *(const v8bf*)(base + 16));
        accs[nt] = wmma_bf16(qf[ks], bfrag, accs[nt]);
      }

    // ---- scale + causal mask + row max (C layout: lane=N col, VGPR j=row) ----
    float pm[8];
#pragma unroll
    for (int j = 0; j < 8; ++j) pm[j] = -1e30f;
    const bool diag = (t == (int)blockIdx.x);
#pragma unroll
    for (int nt = 0; nt < 4; ++nt) {
      const int kcol = kbase + nt * 16 + l16;
#pragma unroll
      for (int j = 0; j < 8; ++j) {
        float s = accs[nt][j] * scale;
        if (diag) {
          const int qrow = qbase + wave * 16 + j + 8 * lhalf;
          if (kcol > qrow) s = -1e30f;
        }
        accs[nt][j] = s;
        pm[j] = fmaxf(pm[j], s);
      }
    }
#pragma unroll
    for (int m = 1; m < 16; m <<= 1)
#pragma unroll
      for (int j = 0; j < 8; ++j) pm[j] = fmaxf(pm[j], __shfl_xor(pm[j], m, 32));

    float alpha[8], rs[8];
#pragma unroll
    for (int j = 0; j < 8; ++j) {
      const float mn = fmaxf(mrow[j], pm[j]);
      alpha[j] = __expf(mrow[j] - mn);
      mrow[j]  = mn;
      rs[j]    = 0.f;
    }
#pragma unroll
    for (int nt = 0; nt < 4; ++nt)
#pragma unroll
      for (int j = 0; j < 8; ++j) {
        const float p = __expf(accs[nt][j] - mrow[j]);
        rs[j] += p;
        sP[(wave * 16 + j + 8 * lhalf) * 64 + nt * 16 + l16] = (__bf16)p;
      }
#pragma unroll
    for (int m = 1; m < 16; m <<= 1)
#pragma unroll
      for (int j = 0; j < 8; ++j) rs[j] += __shfl_xor(rs[j], m, 32);
#pragma unroll
    for (int j = 0; j < 8; ++j) lrow[j] = lrow[j] * alpha[j] + rs[j];
#pragma unroll
    for (int ot = 0; ot < 8; ++ot)
#pragma unroll
      for (int j = 0; j < 8; ++j) acco[ot][j] *= alpha[j];
    __syncthreads();

    // ---- O += P V ----
#pragma unroll
    for (int ks = 0; ks < 2; ++ks) {
      const __bf16* pb = sP + (wave * 16 + l16) * 64 + ks * 32 + lhalf * 8;
      v16bf pf = cat16(*(const v8bf*)pb, *(const v8bf*)(pb + 16));
#pragma unroll
      for (int ot = 0; ot < 8; ++ot) {
        const __bf16* vb = sVt + (ot * 16 + l16) * 64 + ks * 32 + lhalf * 8;
        v16bf vf = cat16(*(const v8bf*)vb, *(const v8bf*)(vb + 16));
        acco[ot] = wmma_bf16(pf, vf, acco[ot]);
      }
    }
    __syncthreads();
  }

#pragma unroll
  for (int ot = 0; ot < 8; ++ot)
#pragma unroll
    for (int j = 0; j < 8; ++j) {
      const int row = qbase + wave * 16 + j + 8 * lhalf;
      const int col = h * 128 + ot * 16 + l16;
      Obf[(size_t)row * 2048 + col] = (__bf16)(acco[ot][j] / lrow[j]);
    }
}

// ---------------------------------------------------------------------------
extern "C" void kernel_launch(void* const* d_in, const int* in_sizes, int n_in,
                              void* d_out, int out_size, void* d_ws, size_t ws_size,
                              hipStream_t stream) {
  (void)in_sizes; (void)n_in; (void)out_size; (void)ws_size;
  const int S = 4096, D = 2048;
  const float scale = 0.088388347648318447f;  // 1/sqrt(128)

  const float* X  = (const float*)d_in[0];
  const float* Wq = (const float*)d_in[1];
  const float* Wk = (const float*)d_in[2];
  const float* Wv = (const float*)d_in[3];
  const float* Wo = (const float*)d_in[4];
  float* out = (float*)d_out;

  __bf16* wsb = (__bf16*)d_ws;
  const size_t SD = (size_t)S * D;
  __bf16* Qh = wsb;            // [H][S][128]
  __bf16* Kh = wsb + SD;
  __bf16* Vh = wsb + 2 * SD;
  __bf16* Ob = wsb + 3 * SD;   // [S][D] row-major

  gemm_proj<<<dim3(D / 128, S / 128, 3), 256, 0, stream>>>(
      X, Wq, Wk, Wv, Qh, Kh, Vh, S, D, D);

  attn_flash<<<dim3(S / 64, 16, 1), 128, 0, stream>>>(Qh, Kh, Vh, Ob, S, scale);

  gemm_out<<<dim3(D / 128, S / 128, 1), 256, 0, stream>>>(Ob, Wo, out, S, D, D);
}
